// HetroGATSum_48928267436267
// MI455X (gfx1250) — compile-verified, matrix operand order checked by reading, not run on hardware
//
#include <hip/hip_runtime.h>

#define N_NODES 50000
#define E_EDGES 640000
#define IN_DIM  64
#define HID     128
#define HEADS   16
#define HDIM    8
#define LAYERS  4
#define TN      64      // output columns per block (4 waves x 16)
#define BSTRIDE 40      // padded k-stride of transposed LDS B tile (16B-aligned)

typedef __attribute__((ext_vector_type(16))) __bf16 v16bf;
typedef __attribute__((ext_vector_type(8)))  __bf16 v8bf;
typedef __attribute__((ext_vector_type(8)))  float  v8f;

// ---------- helpers ----------
static __device__ __forceinline__ __bf16 f2bf(float f) {
    unsigned u = __builtin_bit_cast(unsigned, f);
    unsigned r = u + 0x7FFFu + ((u >> 16) & 1u);   // round-to-nearest-even
    unsigned short s = (unsigned short)(r >> 16);
    return __builtin_bit_cast(__bf16, s);
}

// order-preserving float<->int encoding for atomicMax on signed int
static __device__ __forceinline__ int fenc(float f) {
    int i = __builtin_bit_cast(int, f);
    return i >= 0 ? i : (i ^ 0x7fffffff);
}
static __device__ __forceinline__ float fdec(int i) {
    int j = i >= 0 ? i : (i ^ 0x7fffffff);
    return __builtin_bit_cast(float, j);
}

// ---------- fp32 -> bf16 conversion ----------
__global__ void k_cvt_bf16(const float* __restrict__ src, __bf16* __restrict__ dst, int n)
{
    int i = blockIdx.x * blockDim.x + threadIdx.x;
    if (i < n) dst[i] = f2bf(src[i]);
}

// ---------- WMMA GEMM: C[M x NC] = act(A@B (+ A2@B2) + bias) ----------
// A and B both pre-converted bf16. Block = 4 waves over a 16x64 output strip.
// B chunk staged TRANSPOSED in LDS so each lane's fragment is two contiguous
// b128 DS loads; A fragment is two contiguous b128 global loads. No VALU
// conversion in the hot loop.
__global__ void k_gemm_bf16(const __bf16* __restrict__ A,  int lda,  int K,
                            const __bf16* __restrict__ B,  int ldb,
                            const __bf16* __restrict__ A2, int lda2, int K2,
                            const __bf16* __restrict__ B2, int ldb2,
                            const float* __restrict__ bias,
                            float* __restrict__ C, int ldc,
                            int M, int NC, int act)
{
    __shared__ __bf16 sBt[TN * BSTRIDE];    // 5 KB, [col][k] transposed

    int wave = threadIdx.x >> 5;
    int lane = threadIdx.x & 31;
    int ncg  = NC / TN;
    int row0 = (blockIdx.x / ncg) << 4;
    int colg = (blockIdx.x % ncg) * TN;
    if (row0 >= M) return;                  // uniform per block (grid exact)

    int n16   = lane & 15;
    int khalf = ((lane >> 4) & 1) << 3;     // 0 or 8
    int col   = (wave << 4) + n16;          // this lane's B column within group

    // cooperative staging map: 128 threads x 16 contiguous source bf16
    int tflat = threadIdx.x * 16;
    int srow  = tflat / TN;                 // k row 0..31
    int scol  = tflat % TN;

    v8f c = {0.f, 0.f, 0.f, 0.f, 0.f, 0.f, 0.f, 0.f};

    for (int k0 = 0; k0 < K; k0 += 32) {
        __syncthreads();
        #pragma unroll
        for (int i = 0; i < 16; ++i)        // coalesced read, transposed write
            sBt[(scol + i) * BSTRIDE + srow] = B[(size_t)(k0 + srow) * ldb + colg + scol + i];
        __syncthreads();

        const __bf16* ap = A + (size_t)(row0 + n16) * lda + k0 + khalf;
        v8bf a0 = *(const v8bf*)ap;                 // k = khalf+0..7
        v8bf a1 = *(const v8bf*)(ap + 16);          // k = khalf+16..23
        const __bf16* bp = &sBt[col * BSTRIDE + khalf];
        v8bf b0 = *(const v8bf*)bp;
        v8bf b1 = *(const v8bf*)(bp + 16);
        v16bf a, b;
        #pragma unroll
        for (int i = 0; i < 8; ++i) {
            a[i] = a0[i]; a[i + 8] = a1[i];
            b[i] = b0[i]; b[i + 8] = b1[i];
        }
        c = __builtin_amdgcn_wmma_f32_16x16x32_bf16(false, a, false, b,
                                                    (short)0, c, false, false);
    }
    for (int k0 = 0; k0 < K2; k0 += 32) {
        __syncthreads();
        #pragma unroll
        for (int i = 0; i < 16; ++i)
            sBt[(scol + i) * BSTRIDE + srow] = B2[(size_t)(k0 + srow) * ldb2 + colg + scol + i];
        __syncthreads();

        const __bf16* ap = A2 + (size_t)(row0 + n16) * lda2 + k0 + khalf;
        v8bf a0 = *(const v8bf*)ap;
        v8bf a1 = *(const v8bf*)(ap + 16);
        const __bf16* bp = &sBt[col * BSTRIDE + khalf];
        v8bf b0 = *(const v8bf*)bp;
        v8bf b1 = *(const v8bf*)(bp + 16);
        v16bf a, b;
        #pragma unroll
        for (int i = 0; i < 8; ++i) {
            a[i] = a0[i]; a[i + 8] = a1[i];
            b[i] = b0[i]; b[i + 8] = b1[i];
        }
        c = __builtin_amdgcn_wmma_f32_16x16x32_bf16(false, a, false, b,
                                                    (short)0, c, false, false);
    }

    float bv   = bias ? bias[colg + (wave << 4) + n16] : 0.f;
    int   mrow = ((lane >> 4) & 1) << 3;    // C layout: VGPR j -> M = j + 8*(lane/16)
    #pragma unroll
    for (int j = 0; j < 8; ++j) {
        float v = c[j] + bv;
        if (act == 1) v = v > 0.f ? v : 0.f;
        C[(size_t)(row0 + mrow + j) * ldc + colg + (wave << 4) + n16] = v;
    }
}

// ---------- per-node attention coefficients: el/er[n,h] = dot(z[n,h,:], a[h,:]) ----------
__global__ void k_attn_coef(const float* __restrict__ z,
                            const float* __restrict__ al,
                            const float* __restrict__ ar,
                            float* __restrict__ el, float* __restrict__ er)
{
    int idx = blockIdx.x * blockDim.x + threadIdx.x;   // n*HEADS + h
    if (idx >= N_NODES * HEADS) return;
    int n = idx >> 4, h = idx & 15;
    const float* zp  = z  + (size_t)n * HID + h * HDIM;
    const float* alp = al + h * HDIM;
    const float* arp = ar + h * HDIM;
    float sl = 0.f, sr = 0.f;
    #pragma unroll
    for (int d = 0; d < HDIM; ++d) { float zv = zp[d]; sl += zv * alp[d]; sr += zv * arp[d]; }
    el[idx] = sl; er[idx] = sr;
}

// ---------- init m (encoded -FLT_MAX) and s (0) ----------
__global__ void k_init_ms(int* __restrict__ mE, float* __restrict__ s)
{
    int idx = blockIdx.x * blockDim.x + threadIdx.x;
    if (idx >= N_NODES * HEADS) return;
    mE[idx] = fenc(-3.402823466e38f);
    s[idx]  = 0.f;
}

// ---------- acc[n,c] = b_gat[l,0,c] + b_gat[l,1,c] ----------
__global__ void k_init_acc(float* __restrict__ acc, const float* __restrict__ bg)
{
    int idx = blockIdx.x * blockDim.x + threadIdx.x;
    if (idx >= N_NODES * HID) return;
    int c = idx & (HID - 1);
    acc[idx] = bg[c] + bg[HID + c];
}

// ---------- edge pass 1: segment max ----------
__global__ void k_edge_max(const int* __restrict__ src, const int* __restrict__ dst,
                           const float* __restrict__ el, const float* __restrict__ er,
                           int* __restrict__ mE)
{
    int idx = blockIdx.x * blockDim.x + threadIdx.x;   // e*HEADS + h
    if (idx >= E_EDGES * HEADS) return;
    int e = idx >> 4, h = idx & 15;
    int s = src[e], d = dst[e];
    float v = el[s * HEADS + h] + er[d * HEADS + h];
    v = v >= 0.f ? v : 0.2f * v;                       // leaky 0.2
    atomicMax(&mE[d * HEADS + h], fenc(v));
}

// ---------- edge pass 2: ex = exp(e - m[dst]); s += ex; stash ex ----------
__global__ void k_edge_sumexp(const int* __restrict__ src, const int* __restrict__ dst,
                              const float* __restrict__ el, const float* __restrict__ er,
                              const int* __restrict__ mE,
                              float* __restrict__ ssum, float* __restrict__ exb)
{
    int idx = blockIdx.x * blockDim.x + threadIdx.x;
    if (idx >= E_EDGES * HEADS) return;
    int e = idx >> 4, h = idx & 15;
    int s = src[e], d = dst[e];
    float v = el[s * HEADS + h] + er[d * HEADS + h];
    v = v >= 0.f ? v : 0.2f * v;
    float ex = __expf(v - fdec(mE[d * HEADS + h]));
    exb[idx] = ex;
    atomicAdd(&ssum[d * HEADS + h], ex);
}

// ---------- edge pass 3: acc[dst] += (ex/s[dst]) * z[src] ----------
__global__ void k_edge_scatter(const int* __restrict__ src, const int* __restrict__ dst,
                               const float* __restrict__ exb, const float* __restrict__ ssum,
                               const float* __restrict__ z, float* __restrict__ acc)
{
    int idx = blockIdx.x * blockDim.x + threadIdx.x;
    if (idx >= E_EDGES * HEADS) return;
    int e = idx >> 4, h = idx & 15;
    int s = src[e], d = dst[e];
    float alpha = exb[idx] / ssum[d * HEADS + h];
    const float* zp = z   + (size_t)s * HID + h * HDIM;
    float*       ap = acc + (size_t)d * HID + h * HDIM;
    #pragma unroll
    for (int i = 0; i < HDIM; ++i)
        atomicAdd(&ap[i], alpha * zp[i]);
}

// ---------- h = leaky(acc, 0.01) + h ; also refresh bf16 copy ----------
__global__ void k_layer_fin(const float* __restrict__ acc, float* __restrict__ h,
                            __bf16* __restrict__ hb)
{
    int idx = blockIdx.x * blockDim.x + threadIdx.x;
    if (idx >= N_NODES * HID) return;
    float a = acc[idx];
    float v = h[idx] + (a >= 0.f ? a : 0.01f * a);
    h[idx]  = v;
    hb[idx] = f2bf(v);
}

// ---------- out[n] = dot(t[n,:], W2[:,0]) + b2 : wave-per-node ----------
__global__ void k_dec2(const float* __restrict__ t, const float* __restrict__ W2,
                       const float* __restrict__ b2, float* __restrict__ out)
{
    int n = blockIdx.x * (blockDim.x >> 5) + (threadIdx.x >> 5);
    if (n >= N_NODES) return;
    int lane = threadIdx.x & 31;
    const float* tp = t  + (size_t)n * HID + lane * 4;
    const float* wp = W2 + lane * 4;
    float s = 0.f;
    #pragma unroll
    for (int i = 0; i < 4; ++i) s += tp[i] * wp[i];
    #pragma unroll
    for (int off = 16; off > 0; off >>= 1) s += __shfl_xor(s, off, 32);
    if (lane == 0) out[n] = s + b2[0];
}

// ---------- launcher ----------
extern "C" void kernel_launch(void* const* d_in, const int* in_sizes, int n_in,
                              void* d_out, int out_size, void* d_ws, size_t ws_size,
                              hipStream_t stream)
{
    (void)in_sizes; (void)n_in; (void)out_size; (void)ws_size;

    const float* x       = (const float*)d_in[0];
    const int*   src1    = (const int*)  d_in[1];
    const int*   dst1    = (const int*)  d_in[2];
    const int*   src2    = (const int*)  d_in[3];
    const int*   dst2    = (const int*)  d_in[4];
    const float* eW1     = (const float*)d_in[5];
    const float* eb1     = (const float*)d_in[6];
    const float* eW2     = (const float*)d_in[7];
    const float* eb2     = (const float*)d_in[8];
    const float* eWs     = (const float*)d_in[9];
    const float* W_gat   = (const float*)d_in[10];
    const float* a_l     = (const float*)d_in[11];
    const float* a_r     = (const float*)d_in[12];
    const float* b_gat   = (const float*)d_in[13];
    const float* dW1     = (const float*)d_in[14];
    const float* db1     = (const float*)d_in[15];
    const float* dW2     = (const float*)d_in[16];
    const float* db2     = (const float*)d_in[17];
    float*       out     = (float*)d_out;

    // workspace layout (floats)
    float* ws  = (float*)d_ws;
    float* h   = ws;                                     // N*HID
    float* t   = ws + (size_t)N_NODES * HID;             // N*HID (hidden / z / d)
    float* acc = ws + 2 * (size_t)N_NODES * HID;         // N*HID
    float* el  = ws + 3 * (size_t)N_NODES * HID;         // N*HEADS
    float* er  = el + (size_t)N_NODES * HEADS;           // N*HEADS
    int*   mE  = (int*)(er + (size_t)N_NODES * HEADS);   // N*HEADS
    float* ss  = (float*)(mE + (size_t)N_NODES * HEADS); // N*HEADS
    float* exb = ss + (size_t)N_NODES * HEADS;           // E*HEADS
    // bf16 pool (16B-aligned; each piece a multiple of 8 elems)
    __bf16* bW1 = (__bf16*)(exb + (size_t)E_EDGES * HEADS);
    __bf16* bW2 = bW1 + IN_DIM * HID;
    __bf16* bWs = bW2 + HID * HID;
    __bf16* bG  = bWs + IN_DIM * HID;
    __bf16* bD1 = bG + (size_t)LAYERS * 2 * HID * HID;
    __bf16* xb  = bD1 + HID * HID;                       // N*IN_DIM
    __bf16* tb  = xb + (size_t)N_NODES * IN_DIM;         // N*HID
    __bf16* hb  = tb + (size_t)N_NODES * HID;            // N*HID

    const int gemm_blocks = (N_NODES / 16) * (HID / TN);   // 3125 * 2
    const int nh_blocks   = (N_NODES * HEADS + 255) / 256;
    const int nc_blocks   = (N_NODES * HID + 255) / 256;
    const int nx_blocks   = (N_NODES * IN_DIM + 255) / 256;
    const int eh_blocks   = (E_EDGES * HEADS + 255) / 256;

    // ---- pre-convert weights and x to bf16 ----
    k_cvt_bf16<<<(IN_DIM * HID + 255) / 256, 256, 0, stream>>>(eW1, bW1, IN_DIM * HID);
    k_cvt_bf16<<<(HID * HID + 255) / 256, 256, 0, stream>>>(eW2, bW2, HID * HID);
    k_cvt_bf16<<<(IN_DIM * HID + 255) / 256, 256, 0, stream>>>(eWs, bWs, IN_DIM * HID);
    k_cvt_bf16<<<(LAYERS * 2 * HID * HID + 255) / 256, 256, 0, stream>>>(
        W_gat, bG, LAYERS * 2 * HID * HID);
    k_cvt_bf16<<<(HID * HID + 255) / 256, 256, 0, stream>>>(dW1, bD1, HID * HID);
    k_cvt_bf16<<<nx_blocks, 256, 0, stream>>>(x, xb, N_NODES * IN_DIM);

    // ---- embed MLP with skip ----
    // t = relu(xb @ eW1 + eb1)
    k_gemm_bf16<<<gemm_blocks, 128, 0, stream>>>(
        xb, IN_DIM, IN_DIM, bW1, HID,
        nullptr, 0, 0, nullptr, 0,
        eb1, t, HID, N_NODES, HID, 1);
    k_cvt_bf16<<<nc_blocks, 256, 0, stream>>>(t, tb, N_NODES * HID);
    // h = tb @ eW2 + xb @ eWs + eb2
    k_gemm_bf16<<<gemm_blocks, 128, 0, stream>>>(
        tb, HID, HID, bW2, HID,
        xb, IN_DIM, IN_DIM, bWs, HID,
        eb2, h, HID, N_NODES, HID, 0);
    k_cvt_bf16<<<nc_blocks, 256, 0, stream>>>(h, hb, N_NODES * HID);

    // ---- GAT layers ----
    for (int l = 0; l < LAYERS; ++l) {
        k_init_acc<<<nc_blocks, 256, 0, stream>>>(acc, b_gat + (size_t)l * 2 * HID);
        for (int r = 0; r < 2; ++r) {
            const int* src = (r == 0) ? src1 : src2;
            const int* dst = (r == 0) ? dst1 : dst2;
            const __bf16* W  = bG  + ((size_t)(l * 2 + r)) * HID * HID;
            const float*  al = a_l + ((size_t)(l * 2 + r)) * HEADS * HDIM;
            const float*  ar = a_r + ((size_t)(l * 2 + r)) * HEADS * HDIM;

            // z (=t) = hb @ W
            k_gemm_bf16<<<gemm_blocks, 128, 0, stream>>>(
                hb, HID, HID, W, HID,
                nullptr, 0, 0, nullptr, 0,
                nullptr, t, HID, N_NODES, HID, 0);

            k_attn_coef<<<nh_blocks, 256, 0, stream>>>(t, al, ar, el, er);
            k_init_ms<<<nh_blocks, 256, 0, stream>>>(mE, ss);
            k_edge_max<<<eh_blocks, 256, 0, stream>>>(src, dst, el, er, mE);
            k_edge_sumexp<<<eh_blocks, 256, 0, stream>>>(src, dst, el, er, mE, ss, exb);
            k_edge_scatter<<<eh_blocks, 256, 0, stream>>>(src, dst, exb, ss, t, acc);
        }
        k_layer_fin<<<nc_blocks, 256, 0, stream>>>(acc, h, hb);
    }

    // ---- decision MLP ----
    // t = relu(hb @ dW1 + db1)
    k_gemm_bf16<<<gemm_blocks, 128, 0, stream>>>(
        hb, HID, HID, bD1, HID,
        nullptr, 0, 0, nullptr, 0,
        db1, t, HID, N_NODES, HID, 1);
    // out = t @ dW2 + db2
    k_dec2<<<(N_NODES + 7) / 8, 256, 0, stream>>>(t, dW2, db2, out);
}
